// MultiHeadedAttention_41429254537280
// MI455X (gfx1250) — matrix operand order repaired and checked by reading, not compiled, not run on hardware
//
#include <hip/hip_runtime.h>

#define S_LEN   2048
#define D_MODEL 512
#define NHEAD   8
#define DKK     64
#define BATCH   2
#define MROWS   (BATCH * S_LEN)   // 4096
#define HALFWIN 128

typedef __attribute__((ext_vector_type(16))) __bf16 v16bf;
typedef __attribute__((ext_vector_type(8)))  __bf16 v8bf;
typedef __attribute__((ext_vector_type(8)))  float  v8f;

union Frag16 { v16bf v; v8bf h[2]; };

// ---------------------------------------------------------------------------
// f32 -> bf16 conversion (elementwise)
// ---------------------------------------------------------------------------
__global__ void cvt_bf16_kernel(const float* __restrict__ in,
                                __bf16* __restrict__ out, int n) {
  int i = blockIdx.x * blockDim.x + threadIdx.x;
  if (i < n) out[i] = (__bf16)in[i];
}

// ---------------------------------------------------------------------------
// Wave-tile GEMM, 32x32 output per wave (2x2 register blocking -> 4 WMMAs
// per 32-deep k-step at 32B of fragment loads per WMMA).
// C = A(MxK) @ B(NxK)^T + bias.  A, Bw bf16 row-major. Output routing:
//   0: Q  -> bf16 [B,H,S,DK] row-major
//   1: K  -> bf16 [B,H,S,DK] row-major
//   2: V  -> bf16 [B,H,DK,S] (transposed so PV B-fragments are contiguous)
//   3: f32 row-major MxN (final output projection)
// ---------------------------------------------------------------------------
__global__ __launch_bounds__(128) void gemm_bf16_wmma(
    const __bf16* __restrict__ A, const __bf16* __restrict__ Bw,
    const float* __restrict__ bias, void* __restrict__ out,
    int M, int N, int K, int mode)
{
  const int lane = threadIdx.x & 31;
  const int wave = threadIdx.x >> 5;
  const int half = lane >> 4;      // which 16-lane half
  const int lm   = lane & 15;      // lane within half
  const int ntiles = N >> 5;       // 32-wide tiles

  int tile = blockIdx.x * (blockDim.x >> 5) + wave;
  int mt = tile / ntiles;
  int nt = tile - mt * ntiles;
  int i0 = mt << 5, n0 = nt << 5;
  if (i0 >= M) return;

  const __bf16* arow0 = A  + (size_t)(i0 + lm) * K;
  const __bf16* arow1 = A  + (size_t)(i0 + 16 + lm) * K;
  const __bf16* brow0 = Bw + (size_t)(n0 + lm) * K;
  const __bf16* brow1 = Bw + (size_t)(n0 + 16 + lm) * K;

  v8f acc[2][2] = {};
  for (int k0 = 0; k0 < K; k0 += 32) {
    if (k0 + 32 < K) {
      __builtin_prefetch(arow0 + k0 + 32, 0, 3);
      __builtin_prefetch(arow1 + k0 + 32, 0, 3);
      __builtin_prefetch(brow0 + k0 + 32, 0, 3);
      __builtin_prefetch(brow1 + k0 + 32, 0, 3);
    }
    Frag16 a0, a1, b0, b1;
    // A fragment (16x32): half c holds K = {16c..16c+15}
    a0.v = *(const v16bf*)(arow0 + k0 + 16 * half);
    a1.v = *(const v16bf*)(arow1 + k0 + 16 * half);
    // B fragment (32x16): lane = column n (= row of W); half c holds K = {16c..}
    b0.v = *(const v16bf*)(brow0 + k0 + 16 * half);
    b1.v = *(const v16bf*)(brow1 + k0 + 16 * half);
    acc[0][0] = __builtin_amdgcn_wmma_f32_16x16x32_bf16(false, a0.v, false, b0.v,
                                                        (short)0, acc[0][0], false, false);
    acc[0][1] = __builtin_amdgcn_wmma_f32_16x16x32_bf16(false, a0.v, false, b1.v,
                                                        (short)0, acc[0][1], false, false);
    acc[1][0] = __builtin_amdgcn_wmma_f32_16x16x32_bf16(false, a1.v, false, b0.v,
                                                        (short)0, acc[1][0], false, false);
    acc[1][1] = __builtin_amdgcn_wmma_f32_16x16x32_bf16(false, a1.v, false, b1.v,
                                                        (short)0, acc[1][1], false, false);
  }

  // C layout: VGPR r, lane half c -> M = r + 8c, N = lane%16
  #pragma unroll
  for (int mi = 0; mi < 2; ++mi) {
    #pragma unroll
    for (int ni = 0; ni < 2; ++ni) {
      #pragma unroll
      for (int r = 0; r < 8; ++r) {
        int m  = mi * 16 + r + 8 * half;
        int n  = n0 + ni * 16 + lm;
        int gr = i0 + m;
        float v = acc[mi][ni][r] + bias[n];
        if (mode == 3) {
          ((float*)out)[(size_t)gr * N + n] = v;
        } else {
          __bf16 bv = (__bf16)v;
          int bb = gr >> 11;             // row / 2048
          int ss = gr & (S_LEN - 1);
          int hh = n >> 6;               // n / 64
          int dd = n & (DKK - 1);
          __bf16* o = (__bf16*)out;
          if (mode == 2)
            o[((size_t)(bb * NHEAD + hh) * DKK + dd) * S_LEN + ss] = bv;
          else
            o[((size_t)(bb * NHEAD + hh) * S_LEN + ss) * DKK + dd] = bv;
        }
      }
    }
  }
}

// ---------------------------------------------------------------------------
// total_v[bh][d] = sum_j V[bh][d][j]   (V stored transposed (BH, DK, S))
// ---------------------------------------------------------------------------
__global__ void totalv_kernel(const __bf16* __restrict__ Vt,
                              float* __restrict__ totv) {
  int bh = blockIdx.x;
  int d  = threadIdx.x;
  const __bf16* row = Vt + ((size_t)bh * DKK + d) * S_LEN;
  float s = 0.f;
  for (int j = 0; j < S_LEN; ++j) s += (float)row[j];
  totv[bh * DKK + d] = s;
}

// ---------------------------------------------------------------------------
// Banded attention with exact zero-outside-band softmax rewrite.
// One wave per (b,h, 16-row query tile). 4 waves / block.
//   G_ij = exp(s_ij*scale) - 1 inside band, else 0
//   ctx  = (G @ V + total_v) / (S + rowsum(G))
// Q tile is staged LDS-side with GLOBAL_LOAD_ASYNC_TO_LDS_B128 (ASYNCcnt).
// ---------------------------------------------------------------------------
__global__ __launch_bounds__(128) void attn_band_wmma(
    const __bf16* __restrict__ Q, const __bf16* __restrict__ Kk,
    const __bf16* __restrict__ Vt, const float* __restrict__ totv,
    __bf16* __restrict__ ctx)
{
  __shared__ __attribute__((aligned(16))) __bf16 ldsQ[4][16 * DKK]; // 2KB/wave Q tile
  __shared__ __attribute__((aligned(16))) __bf16 ldsG[4][16 * 32];  // G transpose buf
  __shared__ float ldsDen[4][16];                                   // row sums

  const int lane = threadIdx.x & 31;
  const int wave = threadIdx.x >> 5;
  const int half = lane >> 4;
  const int lm   = lane & 15;

  int wt = blockIdx.x * 4 + wave;       // 0 .. B*H*(S/16)-1 = 2047
  int it = wt & ((S_LEN >> 4) - 1);     // query tile within sequence
  int bh = wt >> 7;                     // b*H + h
  int i0 = it << 4;

  const __bf16* Qb = Q  + (size_t)bh * S_LEN * DKK;
  const __bf16* Kb = Kk + (size_t)bh * S_LEN * DKK;
  const __bf16* Vb = Vt + (size_t)bh * DKK * S_LEN;

  // ---- async-stage the contiguous 16x64 bf16 Q tile (2KB) into LDS
  {
    unsigned ldsBase = (unsigned)(uintptr_t)&ldsQ[wave][0];
    const __bf16* qtile = Qb + (size_t)i0 * DKK;
    #pragma unroll
    for (int t = 0; t < 4; ++t) {
      int chunk = t * 32 + lane;                 // 128 x 16B chunks
      unsigned loff = ldsBase + chunk * 16;
      unsigned long long ga = (unsigned long long)(uintptr_t)(qtile + chunk * 8);
      asm volatile("global_load_async_to_lds_b128 %0, %1, off"
                   :: "v"(loff), "v"(ga) : "memory");
    }
    asm volatile("s_wait_asynccnt 0x0" ::: "memory");
  }

  // Q A-fragments for K-dim halves k0 = 0, 32 (DK = 64), rebuilt from LDS
  Frag16 qa[2];
  {
    const __bf16* qrow = &ldsQ[wave][lm * DKK];
    #pragma unroll
    for (int t = 0; t < 2; ++t) {
      qa[t].v = *(const v16bf*)(qrow + 32 * t + 16 * half);
    }
  }

  v8f accD[4] = {};                     // 16x64 context accumulator
  float rowAcc[8];
  #pragma unroll
  for (int r = 0; r < 8; ++r) rowAcc[r] = 0.f;

  const float scale = 0.125f;           // 1/sqrt(64)
  const int jstart = i0 - HALFWIN;      // band spans 272 cols -> 9 pairs of 16

  for (int p = 0; p < 9; ++p) {
    int jbase = jstart + 32 * p;

    // ---- two 16x16 score tiles -> G (staged to LDS for operand transpose)
    #pragma unroll
    for (int jt = 0; jt < 2; ++jt) {
      int jb = jbase + 16 * jt;
      bool valid = (jb >= 0) && (jb < S_LEN);
      v8f c = {};
      if (valid) {
        const __bf16* krow = Kb + (size_t)(jb + lm) * DKK;
        Frag16 b0, b1;
        b0.v = *(const v16bf*)(krow + 16 * half);
        b1.v = *(const v16bf*)(krow + 32 + 16 * half);
        c = __builtin_amdgcn_wmma_f32_16x16x32_bf16(false, qa[0].v, false, b0.v,
                                                    (short)0, c, false, false);
        c = __builtin_amdgcn_wmma_f32_16x16x32_bf16(false, qa[1].v, false, b1.v,
                                                    (short)0, c, false, false);
      }
      #pragma unroll
      for (int r = 0; r < 8; ++r) {
        int m = r + 8 * half;
        int i = i0 + m;
        int j = jb + lm;
        float g = 0.f;
        if (valid) {
          int dd = i - j; if (dd < 0) dd = -dd;
          if (dd <= HALFWIN) g = __expf(c[r] * scale) - 1.0f;
        }
        // row-sum across the 16 lanes of this half (row m lives in one VGPR/half)
        float t = g;
        t += __shfl_xor(t, 1, 32);
        t += __shfl_xor(t, 2, 32);
        t += __shfl_xor(t, 4, 32);
        t += __shfl_xor(t, 8, 32);
        rowAcc[r] += t;
        ldsG[wave][m * 32 + jt * 16 + lm] = (__bf16)g;
      }
    }

    // ---- G @ V for this 32-wide j slab (skip if fully out of range)
    if (jbase + 31 >= 0 && jbase < S_LEN) {
      Frag16 ga;
      const __bf16* grow = &ldsG[wave][lm * 32];
      ga.v = *(const v16bf*)(grow + 16 * half);

      int jc = jbase + 16 * half;       // K (=j) origin for this half's B chunk
      if (jc < 0) jc = 0;               // clamped loads hit G==0 columns
      if (jc > S_LEN - 16) jc = S_LEN - 16;
      #pragma unroll
      for (int nt = 0; nt < 4; ++nt) {
        Frag16 vb;
        vb.v = *(const v16bf*)(Vb + (size_t)(nt * 16 + lm) * S_LEN + jc);
        accD[nt] = __builtin_amdgcn_wmma_f32_16x16x32_bf16(false, ga.v, false, vb.v,
                                                           (short)0, accD[nt], false, false);
      }
    }
  }

  // ---- publish per-row sums, then finalize
  if (lm == 0) {
    #pragma unroll
    for (int r = 0; r < 8; ++r) ldsDen[wave][r + 8 * half] = rowAcc[r];
  }
  int bb = bh >> 3, hh = bh & 7;
  #pragma unroll
  for (int nt = 0; nt < 4; ++nt) {
    int n = nt * 16 + lm;
    float tv = totv[bh * DKK + n];
    #pragma unroll
    for (int r = 0; r < 8; ++r) {
      int m = r + 8 * half;
      float denom = (float)S_LEN + ldsDen[wave][m];
      float o = (accD[nt][r] + tv) / denom;
      ctx[((size_t)(bb * S_LEN) + i0 + m) * D_MODEL + hh * DKK + n] = (__bf16)o;
    }
  }
}

// ---------------------------------------------------------------------------
// Host launch
// ---------------------------------------------------------------------------
extern "C" void kernel_launch(void* const* d_in, const int* in_sizes, int n_in,
                              void* d_out, int out_size, void* d_ws, size_t ws_size,
                              hipStream_t stream) {
  (void)in_sizes; (void)n_in; (void)out_size; (void)ws_size;
  const float* query  = (const float*)d_in[0];
  const float* key_in = (const float*)d_in[1];
  const float* value  = (const float*)d_in[2];
  const float* Wq = (const float*)d_in[3]; const float* bq = (const float*)d_in[4];
  const float* Wk = (const float*)d_in[5]; const float* bk = (const float*)d_in[6];
  const float* Wv = (const float*)d_in[7]; const float* bv = (const float*)d_in[8];
  const float* Wo = (const float*)d_in[9]; const float* bo = (const float*)d_in[10];

  char* ws = (char*)d_ws;
  size_t off = 0;
  auto alloc = [&](size_t bytes) -> void* {
    void* p = ws + off;
    off = (off + bytes + 255) & ~(size_t)255;
    return p;
  };
  const size_t nX = (size_t)MROWS * D_MODEL;          // 4096*512
  const size_t nW = (size_t)D_MODEL * D_MODEL;        // 512*512
  __bf16* Xq  = (__bf16*)alloc(nX * 2);
  __bf16* Xk  = (__bf16*)alloc(nX * 2);
  __bf16* Xv  = (__bf16*)alloc(nX * 2);
  __bf16* Wqb = (__bf16*)alloc(nW * 2);
  __bf16* Wkb = (__bf16*)alloc(nW * 2);
  __bf16* Wvb = (__bf16*)alloc(nW * 2);
  __bf16* Wob = (__bf16*)alloc(nW * 2);
  __bf16* Qb  = (__bf16*)alloc(nX * 2);               // (B,H,S,DK)
  __bf16* Kb  = (__bf16*)alloc(nX * 2);               // (B,H,S,DK)
  __bf16* Vt  = (__bf16*)alloc(nX * 2);               // (B,H,DK,S)
  float*  totv = (float*)alloc((size_t)BATCH * NHEAD * DKK * 4);
  __bf16* ctx = (__bf16*)alloc(nX * 2);               // (B*S, D)

  // 1) convert to bf16
  cvt_bf16_kernel<<<(int)(nX / 256), 256, 0, stream>>>(query,  Xq, (int)nX);
  cvt_bf16_kernel<<<(int)(nX / 256), 256, 0, stream>>>(key_in, Xk, (int)nX);
  cvt_bf16_kernel<<<(int)(nX / 256), 256, 0, stream>>>(value,  Xv, (int)nX);
  cvt_bf16_kernel<<<(int)(nW / 256), 256, 0, stream>>>(Wq, Wqb, (int)nW);
  cvt_bf16_kernel<<<(int)(nW / 256), 256, 0, stream>>>(Wk, Wkb, (int)nW);
  cvt_bf16_kernel<<<(int)(nW / 256), 256, 0, stream>>>(Wv, Wvb, (int)nW);
  cvt_bf16_kernel<<<(int)(nW / 256), 256, 0, stream>>>(Wo, Wob, (int)nW);

  // 2) QKV projections (WMMA, 32x32 tile per wave)
  const int tiles = (MROWS / 32) * (D_MODEL / 32);    // 2048
  const int gblk  = tiles / 4;                         // 4 waves per block
  gemm_bf16_wmma<<<gblk, 128, 0, stream>>>(Xq, Wqb, bq, Qb, MROWS, D_MODEL, D_MODEL, 0);
  gemm_bf16_wmma<<<gblk, 128, 0, stream>>>(Xk, Wkb, bk, Kb, MROWS, D_MODEL, D_MODEL, 1);
  gemm_bf16_wmma<<<gblk, 128, 0, stream>>>(Xv, Wvb, bv, Vt, MROWS, D_MODEL, D_MODEL, 2);

  // 3) per-(b,h) total value sums
  totalv_kernel<<<BATCH * NHEAD, DKK, 0, stream>>>(Vt, totv);

  // 4) banded attention (WMMA + async-to-LDS staging)
  const int wtiles = BATCH * NHEAD * (S_LEN / 16);     // 2048 wave-tiles
  attn_band_wmma<<<wtiles / 4, 128, 0, stream>>>(Qb, Kb, Vt, totv, ctx);

  // 5) output projection (WMMA, f32 out)
  gemm_bf16_wmma<<<gblk, 128, 0, stream>>>(ctx, Wob, bo, d_out, MROWS, D_MODEL, D_MODEL, 3);
}
